// LSTM_14096082666136
// MI455X (gfx1250) — compile-verified, matrix operand order checked by reading, not run on hardware
//
#include <hip/hip_runtime.h>
#include <math.h>

// ---------------------------------------------------------------------------
// Bidirectional LSTM, persistent-kernel design for MI455X (gfx1250, wave32).
//   B=128, T=2048, IN=128, H=256, G=4H=1024.
// - bf16 WMMA (v_wmma_f32_16x16x32_bf16), fp32 accumulation.
// - W_ih fragments are timestep-invariant -> register-resident (128 VGPRs),
//   removing 1/3 of the per-step LDS weight traffic.
// - Async global->LDS DMA (global_load_async_to_lds_b128, ASYNCcnt) for
//   x_t / h_{t-1} staging; uint4 fallback if the builtin is absent.
// - c-state lives in fp32 registers for the whole scan.
// - 16 WGs per direction; each owns 16 columns of each of i/f/g/o so the
//   pointwise LSTM update is register-local after the WMMAs.
// - Per-timestep grid sync via a per-direction monotonic atomic counter;
//   h exchanged through ping-pong global buffers (L2-resident, 64KB).
// - LDS per WG: 128KB (W_hh 32K + h 64K + x_t 32K).
// ---------------------------------------------------------------------------

typedef __attribute__((ext_vector_type(16))) __bf16 v16bf;
typedef __attribute__((ext_vector_type(8)))  __bf16 v8bf;
typedef __attribute__((ext_vector_type(8)))  float  v8f;
typedef __attribute__((ext_vector_type(4)))  int    vi4;

// address-space-qualified int4 for the async-LDS DMA builtin
typedef __attribute__((address_space(1))) vi4 gvi4;   // global
typedef __attribute__((address_space(3))) vi4 lvi4;   // LDS

#define B_       128
#define T_       2048
#define IN_      128
#define H_       256
#define G_       1024
#define NWG_DIR  16
#define NTHREADS 256

#if __has_builtin(__builtin_amdgcn_global_load_async_to_lds_b128)
#define HAVE_ASYNC_LDS 1
#else
#define HAVE_ASYNC_LDS 0
#endif

__device__ __forceinline__ unsigned short f32_bf16(float f) {
  unsigned u = __float_as_uint(f);
  unsigned r = (u + 0x7FFFu + ((u >> 16) & 1u)) >> 16;  // RNE
  return (unsigned short)r;
}
__device__ __forceinline__ float bf16_f32(unsigned short s) {
  return __uint_as_float(((unsigned)s) << 16);
}
__device__ __forceinline__ float sigm(float x) { return 1.0f / (1.0f + __expf(-x)); }

// 16B global -> LDS stage, async DMA when available
__device__ __forceinline__ void stage16(const unsigned short* g, unsigned short* l) {
#if HAVE_ASYNC_LDS
  __builtin_amdgcn_global_load_async_to_lds_b128((gvi4*)g, (lvi4*)l, 0, 0);
#else
  *(uint4*)l = *(const uint4*)g;
#endif
}
__device__ __forceinline__ void wait_stage() {
#if HAVE_ASYNC_LDS
#if __has_builtin(__builtin_amdgcn_s_wait_asynccnt)
  __builtin_amdgcn_s_wait_asynccnt(0);
#else
  asm volatile("s_wait_asynccnt 0x0" ::: "memory");
#endif
#endif
}

// build a 16-elem bf16 fragment from two 16B halves
__device__ __forceinline__ v16bf mkfrag(const unsigned short* base, int off_lo, int off_hi) {
  v8bf lo = *(const v8bf*)(base + off_lo);
  v8bf hi = *(const v8bf*)(base + off_hi);
  return __builtin_shufflevector(lo, hi, 0,1,2,3,4,5,6,7,8,9,10,11,12,13,14,15);
}

// ---------------- prep: xs fp32 -> bf16 (halves per-step fetch bytes) -------
__global__ void cvt_xs_kernel(const float* __restrict__ src,
                              unsigned short* __restrict__ dst, int n4) {
  int i = blockIdx.x * blockDim.x + threadIdx.x;
  int stride = gridDim.x * blockDim.x;
  for (; i < n4; i += stride) {
    float4 f = ((const float4*)src)[i];
    ushort4 o;
    o.x = f32_bf16(f.x); o.y = f32_bf16(f.y);
    o.z = f32_bf16(f.z); o.w = f32_bf16(f.w);
    ((ushort4*)dst)[i] = o;
  }
}

// ---------------- prep: weights->bf16, h0/c0, h ping-pong init, sync ctr ----
__global__ void prep_kernel(const float* __restrict__ Wih, const float* __restrict__ Whh,
                            const float* __restrict__ h0w, const float* __restrict__ h0b,
                            const float* __restrict__ c0w, const float* __restrict__ c0b,
                            unsigned short* __restrict__ Wih16,
                            unsigned short* __restrict__ Whh16,
                            unsigned short* __restrict__ hbuf,
                            float* __restrict__ h0c0,
                            unsigned int* __restrict__ ctr) {
  int gtid = blockIdx.x * blockDim.x + threadIdx.x;
  int gstride = gridDim.x * blockDim.x;
  for (int i = gtid; i < G_ * IN_; i += gstride) Wih16[i] = f32_bf16(Wih[i]);
  for (int i = gtid; i < G_ * H_;  i += gstride) Whh16[i] = f32_bf16(Whh[i]);
  for (int i = gtid; i < H_; i += gstride) {
    h0c0[i]      = h0w[i] + h0b[i];   // h0_w is (H,1)
    h0c0[H_ + i] = c0w[i] + c0b[i];
  }
  for (int i = gtid; i < B_ * H_; i += gstride) {
    unsigned short v = f32_bf16(h0w[i % H_] + h0b[i % H_]);
    hbuf[i] = v;                      // dir0, buffer 0
    hbuf[2 * B_ * H_ + i] = v;        // dir1, buffer 0
  }
  if (gtid == 0) { ctr[0] = 0u; ctr[1] = 0u; }
}

// ---------------- persistent recurrent kernel -------------------------------
__global__ __launch_bounds__(NTHREADS)
void lstm_persistent(const unsigned short* __restrict__ xs16,
                     const unsigned char*  __restrict__ mask,   // jnp bool = 1 byte
                     const unsigned short* __restrict__ Wih16,
                     const unsigned short* __restrict__ Whh16,
                     const float* __restrict__ bias,
                     const float* __restrict__ h0c0,
                     unsigned short* __restrict__ hbuf,
                     unsigned int* __restrict__ ctr,
                     float* __restrict__ out) {
  // 128KB of LDS per workgroup (CDNA5 WGP: 320KB available)
  __shared__ unsigned short sWhh[64][H_];   // 32KB : 16 cols of each gate, K=H
  __shared__ unsigned short sH[B_][H_];     // 64KB : full h_{t-1} (bf16)
  __shared__ unsigned short sX[B_][IN_];    // 32KB : x_t (bf16)

  const int tid   = threadIdx.x;
  const int wg    = blockIdx.x;
  const int dir   = wg / NWG_DIR;         // 0 = forward scan, 1 = reverse scan
  const int p     = wg % NWG_DIR;         // column-slice owner
  const int wave  = tid >> 5;             // 8 waves: one 16-row M-tile each
  const int lane  = tid & 31;
  const int c16   = lane & 15;
  const int khalf = lane >> 4;

  // ---- stage this WG's W_hh slice into LDS (once) ----
  #pragma unroll
  for (int i = 0; i < 8; ++i) {            // 2048 x 16B chunks of W_hh slice
    int chunk = tid + i * NTHREADS;
    int n = chunk >> 5, k8 = (chunk & 31) * 8;
    int q = n >> 4, c = n & 15;
    stage16(Whh16 + (size_t)(q * H_ + p * 16 + c) * H_ + k8, &sWhh[n][k8]);
  }

  // ---- W_ih B-fragments are t-invariant: keep them in registers ----
  v16bf wihf[4][4];                        // [kk][gate q] : 128 VGPRs
  #pragma unroll
  for (int q = 0; q < 4; ++q) {
    const unsigned short* wrow = Wih16 + (size_t)(q * H_ + p * 16 + c16) * IN_;
    #pragma unroll
    for (int kk = 0; kk < 4; ++kk) {
      const int kb = kk * 32;
      wihf[kk][q] = mkfrag(wrow, kb + khalf * 8, kb + 16 + khalf * 8);
    }
  }

  // per-lane gate biases (one column of each gate)
  float bq[4];
  #pragma unroll
  for (int q = 0; q < 4; ++q) bq[q] = bias[q * H_ + p * 16 + c16];

  // fp32 c-state in registers: 8 batch rows per lane, one h-column
  const int hc = p * 16 + c16;             // owned h column
  float cst[8];
  {
    float c0v = h0c0[H_ + hc];
    #pragma unroll
    for (int r = 0; r < 8; ++r) cst[r] = c0v;
  }
  const int m0 = wave * 16 + khalf * 8;    // first of this lane's 8 D-rows

  for (int t = 0; t < T_; ++t) {
    const int te = dir ? (T_ - 1 - t) : t;

    // ---- phase A: stage x_t and h_{t-1} into LDS (async DMA) ----
    const unsigned short* hread = hbuf + (size_t)(dir * 2 + (t & 1)) * B_ * H_;
    #pragma unroll
    for (int i = 0; i < 8; ++i) {          // x_t: 2048 x 16B chunks
      int chunk = tid + i * NTHREADS;
      int b = chunk >> 4, k8 = (chunk & 15) * 8;
      stage16(xs16 + ((size_t)b * T_ + te) * IN_ + k8, &sX[b][k8]);
    }
    #pragma unroll
    for (int i = 0; i < 16; ++i) {         // h: 4096 x 16B chunks
      int chunk = tid + i * NTHREADS;
      int b = chunk >> 5, k8 = (chunk & 31) * 8;
      stage16(hread + (size_t)b * H_ + k8, &sH[b][k8]);
    }
    if (t + 1 < T_) {                      // prefetch next step's x slice
      const int tn = dir ? (T_ - 2 - t) : (t + 1);
      __builtin_prefetch(xs16 + ((size_t)(tid >> 1) * T_ + tn) * IN_ + (tid & 1) * 64, 0, 1);
    }
    wait_stage();
    __syncthreads();

    // ---- phase B: gates = x@Wih^T + h@Whh^T via bf16 WMMA ----
    v8f acc[4];
    #pragma unroll
    for (int q = 0; q < 4; ++q)
      #pragma unroll
      for (int e = 0; e < 8; ++e) acc[q][e] = 0.0f;

    const int arow = wave * 16 + c16;      // A-matrix row for this lane

    // K = IN (128): A from LDS, B from registers -> no weight LDS traffic
    #pragma unroll
    for (int kk = 0; kk < 4; ++kk) {
      const int kb = kk * 32;
      v16bf a = mkfrag(&sX[arow][0], kb + khalf * 8, kb + 16 + khalf * 8);
      #pragma unroll
      for (int q = 0; q < 4; ++q)
        acc[q] = __builtin_amdgcn_wmma_f32_16x16x32_bf16(
            false, a, false, wihf[kk][q], (short)0, acc[q], false, false);
    }
    // K = H (256): batch the 4 B-frag loads per kk, then 4 back-to-back WMMAs
    #pragma unroll
    for (int kk = 0; kk < 8; ++kk) {
      const int kb = kk * 32;
      v16bf a = mkfrag(&sH[arow][0], kb + khalf * 8, kb + 16 + khalf * 8);
      v16bf bm[4];
      #pragma unroll
      for (int q = 0; q < 4; ++q)
        bm[q] = mkfrag(&sWhh[q * 16 + c16][0], kb + khalf * 8, kb + 16 + khalf * 8);
      #pragma unroll
      for (int q = 0; q < 4; ++q)
        acc[q] = __builtin_amdgcn_wmma_f32_16x16x32_bf16(
            false, a, false, bm[q], (short)0, acc[q], false, false);
    }

    // ---- phase C: register-local LSTM pointwise; write h_t slice ----
    float hold[8]; unsigned char mv[8];
    #pragma unroll
    for (int r = 0; r < 8; ++r) {
      mv[r]   = mask[(size_t)(m0 + r) * T_ + te];
      hold[r] = bf16_f32(sH[m0 + r][hc]);
    }
    unsigned short* hwr = hbuf + (size_t)(dir * 2 + ((t + 1) & 1)) * B_ * H_;
    #pragma unroll
    for (int r = 0; r < 8; ++r) {
      float iv = sigm(acc[0][r] + bq[0]);
      float fv = sigm(acc[1][r] + bq[1]);
      float gv = tanhf(acc[2][r] + bq[2]);
      float ov = sigm(acc[3][r] + bq[3]);
      float cn = fv * cst[r] + iv * gv;
      float hn = ov * tanhf(cn);
      bool m = (mv[r] != 0);
      cst[r] = m ? cn : cst[r];
      float hs = m ? hn : hold[r];
      hwr[(size_t)(m0 + r) * H_ + hc] = f32_bf16(hs);
      if (t == T_ - 1)
        out[(size_t)(m0 + r) * (2 * H_) + dir * H_ + hc] = hs;
    }

    // ---- per-direction grid sync (monotonic counter, 16 WGs) ----
    __threadfence();
    __syncthreads();
    if (tid == 0) {
      __hip_atomic_fetch_add(&ctr[dir], 1u, __ATOMIC_RELEASE, __HIP_MEMORY_SCOPE_AGENT);
      const unsigned int target = (unsigned)(NWG_DIR * (t + 1));
      while (__hip_atomic_load(&ctr[dir], __ATOMIC_ACQUIRE, __HIP_MEMORY_SCOPE_AGENT) < target)
        __builtin_amdgcn_s_sleep(1);
    }
    __syncthreads();
    __threadfence();
  }
}

// ---------------------------------------------------------------------------
extern "C" void kernel_launch(void* const* d_in, const int* in_sizes, int n_in,
                              void* d_out, int out_size, void* d_ws, size_t ws_size,
                              hipStream_t stream) {
  const float*         xs   = (const float*)d_in[0];
  const unsigned char* mask = (const unsigned char*)d_in[1];  // bool array
  const float*         Wih  = (const float*)d_in[2];
  const float*         Whh  = (const float*)d_in[3];
  const float*         b    = (const float*)d_in[4];
  const float*         h0w  = (const float*)d_in[5];
  const float*         h0b  = (const float*)d_in[6];
  const float*         c0w  = (const float*)d_in[7];
  const float*         c0b  = (const float*)d_in[8];

  // workspace layout (needs ~65MB; all offsets 16B-aligned)
  char* ws = (char*)d_ws;
  unsigned short* xs16  = (unsigned short*)(ws);                 // 67,108,864 B
  unsigned short* Wih16 = (unsigned short*)(ws + 67108864);      //    262,144 B
  unsigned short* Whh16 = (unsigned short*)(ws + 67371008);      //    524,288 B
  unsigned short* hbuf  = (unsigned short*)(ws + 67895296);      //    262,144 B (2 dirs x 2 ping-pong x B*H)
  float*          h0c0  = (float*)(ws + 68157440);               //      2,048 B
  unsigned int*   ctr   = (unsigned int*)(ws + 68159488);        //          8 B

  cvt_xs_kernel<<<4096, 256, 0, stream>>>(xs, xs16, (B_ * T_ * IN_) / 4);
  prep_kernel<<<256, 256, 0, stream>>>(Wih, Whh, h0w, h0b, c0w, c0b,
                                       Wih16, Whh16, hbuf, h0c0, ctr);
  lstm_persistent<<<2 * NWG_DIR, NTHREADS, 0, stream>>>(
      xs16, mask, Wih16, Whh16, b, h0c0, hbuf, ctr, (float*)d_out);
}
